// Decoder_21895743275147
// MI455X (gfx1250) — matrix operand order repaired and checked by reading, not compiled, not run on hardware
//
#include <hip/hip_runtime.h>
#include <hip/hip_bf16.h>
#include <math.h>

// ---- problem dims (from reference) ----
#define E        256
#define TWO_E    512
#define FOUR_E   1024
#define EIGHT_E  2048
#define SEQ      512
#define VOCAB    50257
#define SOS_ID   1

typedef __attribute__((ext_vector_type(16))) _Float16 v16h;
typedef __attribute__((ext_vector_type(8)))  float    v8f;

// monotonic float -> sortable u32 (all real floats encode > 0)
__device__ __forceinline__ unsigned int fkey(float f) {
  unsigned int u = __float_as_uint(f);
  return (u & 0x80000000u) ? ~u : (u | 0x80000000u);
}

// ---------------- setup kernels ----------------

// emb16[s,e] = (f16) relu(enc_emb[tokens[s], e])
__global__ void k_embed(const int* __restrict__ tokens,
                        const float* __restrict__ enc_emb,
                        _Float16* __restrict__ emb16) {
  int s = blockIdx.x, e = threadIdx.x;
  float v = enc_emb[(size_t)tokens[s] * E + e];
  emb16[s * E + e] = (_Float16)fmaxf(v, 0.0f);
}

// generic f32 -> f16 conversion (grid-strided)
__global__ void k_cvt(const float* __restrict__ src, _Float16* __restrict__ dst, size_t n) {
  size_t i = (size_t)blockIdx.x * blockDim.x + threadIdx.x;
  size_t stride = (size_t)gridDim.x * blockDim.x;
  for (; i < n; i += stride) dst[i] = (_Float16)src[i];
}

// WMMA GEMM: X[dir] = emb @ Wi[dir].T + b[dir]   (M=512, N=1024, K=256)
// one wave (32 lanes) per 16x16 output tile; 8 K-steps of v_wmma_f32_16x16x32_f16
__global__ __launch_bounds__(32) void k_xproj(const _Float16* __restrict__ emb16,
                                              const _Float16* __restrict__ Wi16f,
                                              const _Float16* __restrict__ Wi16b,
                                              const float* __restrict__ b_f,
                                              const float* __restrict__ b_b,
                                              float* __restrict__ Xf,
                                              float* __restrict__ Xb) {
  const int n0 = blockIdx.x * 16, m0 = blockIdx.y * 16, dir = blockIdx.z;
  const _Float16* W = dir ? Wi16b : Wi16f;
  const float* bias = dir ? b_b : b_f;
  float* X = dir ? Xb : Xf;

  const int lane = threadIdx.x & 31;
  const int mn = lane & 15;      // A row within tile / B column within tile
  const int half = lane >> 4;    // lane-half per ISA 16-bit operand layout
  v8f acc = {};
  for (int k0 = 0; k0 < E; k0 += 32) {
    v16h a, b;
    // A (16x32 f16): lane half 0 -> K[k0..k0+7], K[k0+16..k0+23]; half 1 -> +8
    const _Float16* ap = emb16 + (size_t)(m0 + mn) * E + k0 + half * 8;
#pragma unroll
    for (int i = 0; i < 8; ++i) { a[i] = ap[i]; a[8 + i] = ap[16 + i]; }
    // B (32x16 f16): lane half 0 -> K[k0..k0+7], K[k0+16..k0+23]; half 1 -> +8 (N=lane%16)
    const _Float16* bp = W + (size_t)(n0 + mn) * E + k0 + half * 8;
#pragma unroll
    for (int i = 0; i < 8; ++i) { b[i] = bp[i]; b[8 + i] = bp[16 + i]; }
    acc = __builtin_amdgcn_wmma_f32_16x16x32_f16(false, a, false, b, (short)0, acc,
                                                 false, false);
  }
  const float bv = bias[n0 + mn];
  // D layout: lanes 0-15 -> M=r, N=lane ; lanes 16-31 -> M=8+r, N=lane-16
#pragma unroll
  for (int r = 0; r < 8; ++r)
    X[(size_t)(m0 + half * 8 + r) * FOUR_E + n0 + mn] = acc[r] + bv;
}

// sequential BiLSTM scan, one workgroup (1024 thr) per direction
__global__ __launch_bounds__(1024) void k_encscan(const float* __restrict__ Xf,
                                                  const float* __restrict__ Xb,
                                                  const float* __restrict__ Wh_f,
                                                  const float* __restrict__ Wh_b,
                                                  const int* __restrict__ tokens,
                                                  float* __restrict__ encm,
                                                  float* __restrict__ enchid) {
  const int dir = blockIdx.x;
  const float* X = dir ? Xb : Xf;
  const float* Wh = dir ? Wh_b : Wh_f;
  __shared__ float sh[E];
  __shared__ float sg[FOUR_E];
  const int tid = threadIdx.x;
  float c = 0.0f;
  if (tid < E) sh[tid] = 0.0f;
  __syncthreads();
  for (int t = 0; t < SEQ; ++t) {
    const int row = dir ? (SEQ - 1 - t) : t;           // backward consumes emb reversed
    float g = X[(size_t)row * FOUR_E + tid];           // input proj + bias precomputed
    const float* wr = Wh + (size_t)tid * E;
#pragma unroll 4
    for (int k = 0; k < E; ++k) g += wr[k] * sh[k];
    sg[tid] = g;
    __syncthreads();
    if (tid < E) {
      const float ig = sg[tid], fg = sg[E + tid], gg = sg[2 * E + tid], og = sg[3 * E + tid];
      const float si = 1.0f / (1.0f + __expf(-ig));
      const float sf = 1.0f / (1.0f + __expf(-fg));
      const float so = 1.0f / (1.0f + __expf(-og));
      c = sf * c + si * tanhf(gg);
      const float h = so * tanhf(c);
      sh[tid] = h;
      const float mk = (tokens[row] > 0) ? 1.0f : 0.0f;
      encm[(size_t)row * TWO_E + dir * E + tid] = h * mk;  // masked encodings
    }
    __syncthreads();
  }
  if (tid < E) enchid[dir * E + tid] = sh[tid];
}

// decpre[j] = dec_b[j] + dec_Wi[j, 2E:4E] . enc_hidden ; also h0=enc_hidden, c0=0
__global__ void k_decpre(const float* __restrict__ dec_Wi, const float* __restrict__ dec_b,
                         const float* __restrict__ enchid, float* __restrict__ decpre,
                         float* __restrict__ hstate, float* __restrict__ cstate) {
  const int j = blockIdx.x * blockDim.x + threadIdx.x;
  if (j < EIGHT_E) {
    const float* wr = dec_Wi + (size_t)j * FOUR_E + TWO_E;
    float a = dec_b[j];
#pragma unroll 4
    for (int k = 0; k < TWO_E; ++k) a += wr[k] * enchid[k];
    decpre[j] = a;
  }
  if (j < TWO_E) { hstate[j] = enchid[j]; cstate[j] = 0.0f; }
}

__global__ void k_init(unsigned long long* __restrict__ slots, int* __restrict__ words,
                       float* __restrict__ out_sentence) {
  const int i = blockIdx.x * blockDim.x + threadIdx.x;
  if (i < SEQ) slots[i] = 0ull;
  if (i == 0) { words[0] = SOS_ID; out_sentence[0] = (float)SOS_ID; }
}

// ---------------- per-decode-step kernels ----------------

// LSTM cell + attention + W projection, single 1024-thread workgroup
__global__ __launch_bounds__(1024) void k_deccell(
    int t, const int* __restrict__ words, const float* __restrict__ dec_emb,
    const float* __restrict__ dec_Wi, const float* __restrict__ dec_Wh,
    const float* __restrict__ decpre, const float* __restrict__ attn_W,
    const float* __restrict__ attn_b, const float* __restrict__ W_W,
    const float* __restrict__ W_b, const float* __restrict__ encm,
    const int* __restrict__ tokens, float* __restrict__ hstate,
    float* __restrict__ cstate, _Float16* __restrict__ a1_16) {
  __shared__ float se[TWO_E], sh[TWO_E], sq[TWO_E], sw[SEQ], sctx[TWO_E], red[SEQ];
  __shared__ float sg[EIGHT_E];
  const int tid = threadIdx.x;
  const int word = words[t];
  if (tid < TWO_E) {
    sh[tid] = hstate[tid];
    se[tid] = fmaxf(dec_emb[(size_t)word * TWO_E + tid], 0.0f);  // relu(embedding)
  }
  __syncthreads();
  // 2048 gate pre-activations; enc-hidden part + bias folded into decpre
#pragma unroll
  for (int jj = 0; jj < 2; ++jj) {
    const int j = tid + jj * 1024;
    float g = decpre[j];
    const float* wi = dec_Wi + (size_t)j * FOUR_E;  // first 2E columns act on e
    const float* wh = dec_Wh + (size_t)j * TWO_E;
#pragma unroll 4
    for (int k = 0; k < TWO_E; ++k) g += wi[k] * se[k] + wh[k] * sh[k];
    sg[j] = g;
  }
  __syncthreads();
  if (tid < TWO_E) {
    const float ig = sg[tid], fg = sg[TWO_E + tid], gg = sg[2 * TWO_E + tid],
                og = sg[3 * TWO_E + tid];
    const float si = 1.0f / (1.0f + __expf(-ig));
    const float sf = 1.0f / (1.0f + __expf(-fg));
    const float so = 1.0f / (1.0f + __expf(-og));
    const float c = sf * cstate[tid] + si * tanhf(gg);
    const float h = so * tanhf(c);
    cstate[tid] = c; hstate[tid] = h; sh[tid] = h;
  }
  __syncthreads();
  if (tid < TWO_E) {  // q = attn_W @ h + attn_b
    float q = attn_b[tid];
    const float* ar = attn_W + (size_t)tid * TWO_E;
#pragma unroll 4
    for (int k = 0; k < TWO_E; ++k) q += ar[k] * sh[k];
    sq[tid] = q;
  }
  __syncthreads();
  if (tid < SEQ) {  // scores = enc_masked @ q, masked -inf
    float s = -INFINITY;
    if (tokens[tid] > 0) {
      s = 0.0f;
      const float* er = encm + (size_t)tid * TWO_E;
#pragma unroll 4
      for (int k = 0; k < TWO_E; ++k) s += er[k] * sq[k];
    }
    sw[tid] = s; red[tid] = s;
  }
  __syncthreads();
  for (int off = SEQ / 2; off > 0; off >>= 1) {
    if (tid < off) red[tid] = fmaxf(red[tid], red[tid + off]);
    __syncthreads();
  }
  const float mx = red[0];
  __syncthreads();
  if (tid < SEQ) {
    const float s = sw[tid];
    const float wv = (s == -INFINITY) ? 0.0f : __expf(s - mx);
    sw[tid] = wv; red[tid] = wv;
  }
  __syncthreads();
  for (int off = SEQ / 2; off > 0; off >>= 1) {
    if (tid < off) red[tid] += red[tid + off];
    __syncthreads();
  }
  const float sum = red[0];
  __syncthreads();
  const float inv = (sum > 0.0f) ? 1.0f / sum : 0.0f;  // nan->0 path
  if (tid < SEQ) sw[tid] *= inv;
  __syncthreads();
  if (tid < TWO_E) {  // context = w @ enc_masked  (coalesced across columns)
    float cx = 0.0f;
    for (int s = 0; s < SEQ; ++s) cx += sw[s] * encm[(size_t)s * TWO_E + tid];
    sctx[tid] = cx;
  }
  __syncthreads();
  {  // a1 = W_W @ [h, ctx] + W_b, stored f16 for the WMMA logits GEMV
    float a = W_b[tid];
    const float* wr = W_W + (size_t)tid * FOUR_E;
#pragma unroll 4
    for (int k = 0; k < TWO_E; ++k) a += wr[k] * sh[k] + wr[TWO_E + k] * sctx[k];
    a1_16[tid] = (_Float16)a;
  }
}

// logits GEMV via WMMA: each wave owns a 16-row tile of V; K-loop 1024/32 = 32 WMMAs.
// A = lin_W f16 tile (the L2-resident 103MB operand, fully utilized);
// B = a1 broadcast into all 16 columns (LDS); result read from D column 0.
__global__ __launch_bounds__(256) void k_logits(int t, const _Float16* __restrict__ linW16,
                                                const float* __restrict__ lin_b,
                                                const _Float16* __restrict__ a1g,
                                                float* __restrict__ out_logits,
                                                unsigned long long* __restrict__ slot) {
  __shared__ __align__(32) _Float16 sA1[FOUR_E];
  __shared__ unsigned long long skey[256];
  const int tid = threadIdx.x;
  for (int i = tid; i < FOUR_E; i += 256) sA1[i] = a1g[i];
  __syncthreads();

  const int wave = tid >> 5, lane = tid & 31;
  const int tile = blockIdx.x * 8 + wave;
  const int v0 = tile * 16;
  unsigned long long best = 0ull;
  if (v0 < VOCAB) {
    const int mn = lane & 15, half = lane >> 4;
    int arow = v0 + mn;
    if (arow >= VOCAB) arow = VOCAB - 1;  // tail tile: clamp loads, guard stores
    const _Float16* abase = linW16 + (size_t)arow * FOUR_E + half * 8;
    v8f acc = {};
    for (int k0 = 0; k0 < FOUR_E; k0 += 32) {
      if (k0 + 64 < FOUR_E) __builtin_prefetch((const void*)(abase + k0 + 64), 0, 1);
      v16h a, b;
      const _Float16* ap = abase + k0;
#pragma unroll
      for (int i = 0; i < 8; ++i) { a[i] = ap[i]; a[8 + i] = ap[16 + i]; }
#pragma unroll
      for (int i = 0; i < 8; ++i) { b[i] = sA1[k0 + half * 8 + i]; b[8 + i] = sA1[k0 + 16 + half * 8 + i]; }  // all 16 B-columns = a1 chunk
      acc = __builtin_amdgcn_wmma_f32_16x16x32_f16(false, a, false, b, (short)0, acc,
                                                   false, false);
    }
    if (mn == 0) {  // column 0 holders: lane 0 -> M 0..7, lane 16 -> M 8..15
      const int mbase = half * 8;
#pragma unroll
      for (int r = 0; r < 8; ++r) {
        const int v = v0 + mbase + r;
        if (v < VOCAB) {
          const float y = acc[r] + lin_b[v];
          out_logits[v] = y;
          const unsigned long long key =
              ((unsigned long long)fkey(y) << 32) |
              (unsigned long long)(0xFFFFFFFFu - (unsigned)v);  // first-max tiebreak
          if (key > best) best = key;
        }
      }
    }
  }
  skey[tid] = best;
  __syncthreads();
  for (int off = 128; off > 0; off >>= 1) {
    if (tid < off) { if (skey[tid + off] > skey[tid]) skey[tid] = skey[tid + off]; }
    __syncthreads();
  }
  if (tid == 0) atomicMax(slot, skey[0]);
}

__global__ void k_argfin(int t, const unsigned long long* __restrict__ slots,
                         int* __restrict__ words, float* __restrict__ out_sentence) {
  if (threadIdx.x == 0) {
    const unsigned long long key = slots[t];
    const unsigned int v = 0xFFFFFFFFu - (unsigned int)(key & 0xFFFFFFFFull);
    words[t + 1] = (int)v;
    out_sentence[1 + t] = (float)v;
  }
}

// ---------------- host driver ----------------

extern "C" void kernel_launch(void* const* d_in, const int* in_sizes, int n_in,
                              void* d_out, int out_size, void* d_ws, size_t ws_size,
                              hipStream_t stream) {
  const int*   tokens   = (const int*)d_in[0];
  const float* enc_emb  = (const float*)d_in[2];
  const float* dec_emb  = (const float*)d_in[3];
  const float* enc_Wi_f = (const float*)d_in[4];
  const float* enc_Wh_f = (const float*)d_in[5];
  const float* enc_b_f  = (const float*)d_in[6];
  const float* enc_Wi_b = (const float*)d_in[7];
  const float* enc_Wh_b = (const float*)d_in[8];
  const float* enc_b_b  = (const float*)d_in[9];
  const float* dec_Wi   = (const float*)d_in[10];
  const float* dec_Wh   = (const float*)d_in[11];
  const float* dec_b    = (const float*)d_in[12];
  const float* attn_W   = (const float*)d_in[13];
  const float* attn_b   = (const float*)d_in[14];
  const float* W_W      = (const float*)d_in[15];
  const float* W_b      = (const float*)d_in[16];
  const float* lin_W    = (const float*)d_in[17];
  const float* lin_b    = (const float*)d_in[18];
  float* out = (float*)d_out;
  (void)in_sizes; (void)n_in; (void)out_size; (void)ws_size;

  // workspace carve-up (~110 MB; dominated by f16 lin_W which fits in the 192MB L2)
  char* ws = (char*)d_ws;
  size_t off = 0;
  auto alloc = [&](size_t bytes) -> char* {
    char* p = ws + off;
    off += (bytes + 255) & ~(size_t)255;
    return p;
  };
  _Float16* linW16 = (_Float16*)alloc((size_t)VOCAB * FOUR_E * sizeof(_Float16));
  _Float16* emb16  = (_Float16*)alloc((size_t)SEQ * E * sizeof(_Float16));
  _Float16* Wi16f  = (_Float16*)alloc((size_t)FOUR_E * E * sizeof(_Float16));
  _Float16* Wi16b  = (_Float16*)alloc((size_t)FOUR_E * E * sizeof(_Float16));
  float* Xf     = (float*)alloc((size_t)SEQ * FOUR_E * sizeof(float));
  float* Xb     = (float*)alloc((size_t)SEQ * FOUR_E * sizeof(float));
  float* encm   = (float*)alloc((size_t)SEQ * TWO_E * sizeof(float));
  float* enchid = (float*)alloc(TWO_E * sizeof(float));
  float* hstate = (float*)alloc(TWO_E * sizeof(float));
  float* cstate = (float*)alloc(TWO_E * sizeof(float));
  float* decpre = (float*)alloc(EIGHT_E * sizeof(float));
  _Float16* a116 = (_Float16*)alloc(FOUR_E * sizeof(_Float16));
  unsigned long long* slots = (unsigned long long*)alloc(SEQ * sizeof(unsigned long long));
  int* words = (int*)alloc((SEQ + 1) * sizeof(int));

  // setup
  k_embed<<<SEQ, E, 0, stream>>>(tokens, enc_emb, emb16);
  k_cvt<<<2048, 256, 0, stream>>>(lin_W, linW16, (size_t)VOCAB * FOUR_E);
  k_cvt<<<512, 256, 0, stream>>>(enc_Wi_f, Wi16f, (size_t)FOUR_E * E);
  k_cvt<<<512, 256, 0, stream>>>(enc_Wi_b, Wi16b, (size_t)FOUR_E * E);
  k_xproj<<<dim3(FOUR_E / 16, SEQ / 16, 2), 32, 0, stream>>>(emb16, Wi16f, Wi16b,
                                                             enc_b_f, enc_b_b, Xf, Xb);
  k_encscan<<<2, 1024, 0, stream>>>(Xf, Xb, enc_Wh_f, enc_Wh_b, tokens, encm, enchid);
  k_decpre<<<EIGHT_E / 256, 256, 0, stream>>>(dec_Wi, dec_b, enchid, decpre, hstate, cstate);
  k_init<<<2, 256, 0, stream>>>(slots, words, out);

  // autoregressive decode: 512 dependent steps, stream-serialized
  const int ntiles = (VOCAB + 15) / 16;       // 3142
  const int nblocks = (ntiles + 7) / 8;       // 393
  float* logits_base = out + (SEQ + 1);
  for (int t = 0; t < SEQ; ++t) {
    k_deccell<<<1, 1024, 0, stream>>>(t, words, dec_emb, dec_Wi, dec_Wh, decpre, attn_W,
                                      attn_b, W_W, W_b, encm, tokens, hstate, cstate, a116);
    k_logits<<<nblocks, 256, 0, stream>>>(t, linW16, lin_b, a116,
                                          logits_base + (size_t)t * VOCAB, slots + t);
    k_argfin<<<1, 32, 0, stream>>>(t, slots, words, out);
  }
}
